// MaskedModel_82025285419746
// MI455X (gfx1250) — compile-verified, matrix operand order checked by reading, not run on hardware
//
#include <hip/hip_runtime.h>
#include <hip/hip_bf16.h>

// Problem geometry
#define TOTAL 3310   // 2500+500+200+100+10
#define KPAD  3328   // padded: multiple of 32 (K-step) and 256 (N tile)
#define BATCH 8192
#define NX    2500

typedef __attribute__((ext_vector_type(16))) __bf16 v16bf;
typedef __attribute__((ext_vector_type(8)))  __bf16 v8bf;
typedef __attribute__((ext_vector_type(8)))  float  v8f;
typedef __attribute__((ext_vector_type(4)))  int    v4i;

#if __has_builtin(__builtin_amdgcn_global_load_async_to_lds_b128)
#define USE_ASYNC 1
#else
#define USE_ASYNC 0
#endif

typedef __attribute__((address_space(3))) char*        lds_char_p;
typedef __attribute__((address_space(3))) v8bf*        lds_v8bf_p;
typedef __attribute__((address_space(3))) const v8bf*  lds_v8bf_cp;
typedef __attribute__((address_space(3))) v4i*         lds_v4i_p;
typedef __attribute__((address_space(1))) v4i*         glob_v4i_p;

// Generic LDS pointer -> AS3: generic LDS addresses carry the LDS offset in
// addr[31:0] (ISA §10.2 aperture mapping), so truncation is the addrspacecast.
__device__ __forceinline__ lds_char_p to_lds(void* p) {
  return (lds_char_p)(unsigned int)(unsigned long long)p;
}

__device__ __forceinline__ void wait_async_all() {
#if USE_ASYNC
#if __has_builtin(__builtin_amdgcn_s_wait_asynccnt)
  __builtin_amdgcn_s_wait_asynccnt(0);
#else
  asm volatile("s_wait_asynccnt 0x0" ::: "memory");
#endif
#endif
}

// Copy one 16B chunk global -> LDS (async on CDNA5; sync ds_store fallback).
__device__ __forceinline__ void stage16(const __bf16* g, lds_char_p l) {
#if USE_ASYNC
  __builtin_amdgcn_global_load_async_to_lds_b128(
      (glob_v4i_p)(unsigned long long)g, (lds_v4i_p)l, 0, 0);
#else
  *(lds_v8bf_p)l = *(const v8bf*)g;
#endif
}

// Fragment-ordered LDS address for tile chunk (row r, 16B-chunk c):
// fragment block = (r/16)*1KB; ds_load0 region c∈{0,1}, ds_load1 region c∈{2,3};
// within a region lanes 0-15 get c even (rows), lanes 16-31 get c odd.
__device__ __forceinline__ int chunk_off(int r, int c) {
  return ((r >> 4) << 10) + ((c >> 1) << 9) + ((c & 1) << 8) + ((r & 15) << 4);
}

// Stage one K-step: A tile 128x32 (8KB, 2 chunks/thread) + B tile 256x32
// (16KB at +8192, 4 chunks/thread) into `buf`, scattered to fragment order.
__device__ __forceinline__ void stage_tiles(const __bf16* A, const __bf16* Bt,
                                            int wgM0, int wgN0, int k0,
                                            lds_char_p buf, int tid) {
#pragma unroll
  for (int i = 0; i < 2; ++i) {
    int q = tid + i * 256, r = q >> 2, c = q & 3;
    stage16(A + (size_t)(wgM0 + r) * KPAD + k0 + c * 8, buf + chunk_off(r, c));
  }
#pragma unroll
  for (int i = 0; i < 4; ++i) {
    int q = tid + i * 256, r = q >> 2, c = q & 3;
    stage16(Bt + (size_t)(wgN0 + r) * KPAD + k0 + c * 8,
            buf + 8192 + chunk_off(r, c));
  }
}

__device__ __forceinline__ v16bf cat16(v8bf lo, v8bf hi) {
  return __builtin_shufflevector(lo, hi, 0, 1, 2, 3, 4, 5, 6, 7,
                                 8, 9, 10, 11, 12, 13, 14, 15);
}

// ---------- prep kernels ----------
__global__ void prep_wt(const float* __restrict__ W, const float* __restrict__ M,
                        __bf16* __restrict__ Wt) {
  int k = blockIdx.x * 256 + threadIdx.x;
  int n = blockIdx.y;
  if (k >= KPAD) return;
  float v = 0.f;
  if (k < TOTAL && n < TOTAL) v = W[(size_t)k * TOTAL + n] * M[(size_t)k * TOTAL + n];
  Wt[(size_t)n * KPAD + k] = (__bf16)v;
}

__global__ void prep_h0(const float* __restrict__ X, __bf16* __restrict__ H0) {
  int col = blockIdx.x * 256 + threadIdx.x;
  int row = blockIdx.y;
  if (col >= KPAD) return;
  float v = (col < NX) ? X[(size_t)row * NX + col] : 0.f;
  H0[(size_t)row * KPAD + col] = (__bf16)v;
}

__global__ void prep_bias(const float* __restrict__ b, float* __restrict__ bp) {
  int i = blockIdx.x * 256 + threadIdx.x;
  if (i < KPAD) bp[i] = (i < TOTAL) ? b[i] : 0.f;
}

// ---------- GEMM: C[M,N] = A[M,K] @ Bt[N,K]^T + bias ----------
// FINAL==0: store relu(C) bf16 (ld=KPAD).  FINAL==1: store C f32 (ld=TOTAL).
template <int FINAL>
__global__ __launch_bounds__(256)
void gemm_wmma(const __bf16* __restrict__ A, const __bf16* __restrict__ Bt,
               const float* __restrict__ bias,
               __bf16* __restrict__ Obf, float* __restrict__ Of32) {
  __shared__ char smem[2 * 24576];  // double-buffered [A 8KB | B 16KB]
  lds_char_p lds = to_lds(smem);

  const int tid = threadIdx.x;
  const int wave = tid >> 5, lane = tid & 31, lane16 = lane & 15;
  const int wgM0 = blockIdx.y * 128, wgN0 = blockIdx.x * 256;
  const int wm0 = wgM0 + (wave >> 2) * 64;   // wave M origin
  const int wn0 = wgN0 + (wave & 3) * 64;    // wave N origin
  const int faB = (wave >> 2) * 4;           // wave's first A fragment index
  const int fbB = (wave & 3) * 4;            // wave's first B fragment index

  v8f acc[4][4];
#pragma unroll
  for (int nf = 0; nf < 4; ++nf) {
    float bv = bias[wn0 + nf * 16 + lane16];
#pragma unroll
    for (int mf = 0; mf < 4; ++mf)
#pragma unroll
      for (int r = 0; r < 8; ++r) acc[mf][nf][r] = bv;
  }

  const int NK = KPAD / 32;  // 104
  stage_tiles(A, Bt, wgM0, wgN0, 0, lds, tid);
  wait_async_all();
  __syncthreads();

  for (int step = 0; step < NK; ++step) {
    lds_char_p cur = lds + (step & 1) * 24576;
    if (step + 1 < NK)
      stage_tiles(A, Bt, wgM0, wgN0, (step + 1) * 32,
                  lds + ((step + 1) & 1) * 24576, tid);

    v16bf afrag[4], bfrag[4];
#pragma unroll
    for (int mf = 0; mf < 4; ++mf) {
      lds_char_p p = cur + (faB + mf) * 1024 + lane * 16;
      afrag[mf] = cat16(*(lds_v8bf_cp)p, *(lds_v8bf_cp)(p + 512));
    }
#pragma unroll
    for (int nf = 0; nf < 4; ++nf) {
      lds_char_p p = cur + 8192 + (fbB + nf) * 1024 + lane * 16;
      bfrag[nf] = cat16(*(lds_v8bf_cp)p, *(lds_v8bf_cp)(p + 512));
    }
#pragma unroll
    for (int mf = 0; mf < 4; ++mf)
#pragma unroll
      for (int nf = 0; nf < 4; ++nf)
        acc[mf][nf] = __builtin_amdgcn_wmma_f32_16x16x32_bf16(
            false, afrag[mf], false, bfrag[nf], (short)0, acc[mf][nf],
            false, false);

    wait_async_all();   // next-buffer fills landed
    __syncthreads();    // everyone done reading cur / writing next
  }

  // store: C layout — lane<16: N=lane16, M=r; lane>=16: M=r+8
#pragma unroll
  for (int mf = 0; mf < 4; ++mf) {
#pragma unroll
    for (int nf = 0; nf < 4; ++nf) {
      const int col = wn0 + nf * 16 + lane16;
#pragma unroll
      for (int r = 0; r < 8; ++r) {
        const int row = wm0 + mf * 16 + r + ((lane < 16) ? 0 : 8);
        float v = acc[mf][nf][r];
        if (FINAL) {
          if (col < TOTAL) Of32[(size_t)row * TOTAL + col] = v;
        } else {
          Obf[(size_t)row * KPAD + col] = (__bf16)(v > 0.f ? v : 0.f);
        }
      }
    }
  }
}

extern "C" void kernel_launch(void* const* d_in, const int* in_sizes, int n_in,
                              void* d_out, int out_size, void* d_ws, size_t ws_size,
                              hipStream_t stream) {
  const float* X  = (const float*)d_in[0];
  const float* W  = (const float*)d_in[1];
  const float* b  = (const float*)d_in[2];
  const float* Bm = (const float*)d_in[3];

  const size_t WT_BYTES   = (size_t)KPAD * KPAD * sizeof(__bf16);   // ~22.2 MB
  const size_t BIAS_BYTES = (size_t)KPAD * sizeof(float);
  const size_t H_BYTES    = (size_t)BATCH * KPAD * sizeof(__bf16);  // ~54.5 MB
  char* ws = (char*)d_ws;
  __bf16* Wt = (__bf16*)(ws);
  float*  bp = (float*)(ws + WT_BYTES);
  __bf16* hA = (__bf16*)(ws + WT_BYTES + BIAS_BYTES);
  __bf16* hB = (__bf16*)(ws + WT_BYTES + BIAS_BYTES + H_BYTES);
  float*  out = (float*)d_out;

  prep_wt  <<<dim3(KPAD / 256, KPAD), 256, 0, stream>>>(W, Bm, Wt);
  prep_h0  <<<dim3(KPAD / 256, BATCH), 256, 0, stream>>>(X, hA);
  prep_bias<<<dim3(KPAD / 256), 256, 0, stream>>>(b, bp);

  dim3 grid(KPAD / 256, BATCH / 128), blk(256);
  gemm_wmma<0><<<grid, blk, 0, stream>>>(hA, Wt, bp, hB, nullptr);  // h1
  gemm_wmma<0><<<grid, blk, 0, stream>>>(hB, Wt, bp, hA, nullptr);  // h2
  gemm_wmma<0><<<grid, blk, 0, stream>>>(hA, Wt, bp, hB, nullptr);  // h3
  gemm_wmma<1><<<grid, blk, 0, stream>>>(hB, Wt, bp, nullptr, out); // h4
}
